// GPTQMarlinFP8Linear_65738769433006
// MI455X (gfx1250) — compile-verified, matrix operand order checked by reading, not run on hardware
//
#include <hip/hip_runtime.h>
#include <hip/hip_bf16.h>

typedef __attribute__((ext_vector_type(16))) _Float16 v16h;
typedef __attribute__((ext_vector_type(8)))  _Float16 v8h;
typedef __attribute__((ext_vector_type(2)))  _Float16 v2h;
typedef __attribute__((ext_vector_type(8)))  float    v8f;
typedef __attribute__((ext_vector_type(2)))  float    v2f;

#define M_DIM   32
#define N_DIM   12288
#define K_DIM   4096
#define KSPLIT  8
#define FP8_MAXF 448.0f
#define WS_A16_OFFSET 256   // byte offset of f16 A copy inside d_ws

#if __has_builtin(__builtin_amdgcn_cvt_pk_fp8_f32) && __has_builtin(__builtin_amdgcn_cvt_pk_f32_fp8)
#define HW_FP8 1
#endif
#if defined(HW_FP8) && __has_builtin(__builtin_amdgcn_cvt_pk_f16_fp8)
#define HW_FP8_CVT16 1
#endif

__device__ __forceinline__ float clamp448(float y) {
    return fminf(fmaxf(y, -FP8_MAXF), FP8_MAXF);   // -> v_med3_num_f32
}

// Software fallback: RNE quantize to e4m3 grid (fp8-space value), branchless.
__device__ __forceinline__ float e4m3_rne_sw(float y) {
    y = clamp448(y);
    float a = fabsf(y);
    unsigned u = __float_as_uint(a);
    unsigned r = (u + ((u >> 20) & 1u) + 0x7FFFFu) & 0xFFF00000u;
    float an = __uint_as_float(r);                  // normal: RNE to 3 mantissa bits
    float as = rintf(a * 512.0f) * 0.001953125f;    // subnormal: grid of 2^-9
    float q = (a >= 0.015625f) ? an : as;           // v_cndmask
    return copysignf(q, y);
}

// Quantize 4 scaled values to e4m3 (RNE), write 4 fp8-space f16 to dst[base..].
// `p` is a persistent pack register carried across iterations so the 16-bit
// partial writes always merge into an already-defined register (no zero movs).
__device__ __forceinline__ void e4m3_quad(float y0, float y1, float y2, float y3,
                                          int& p, v16h& dst, int base) {
#ifdef HW_FP8
    p = __builtin_amdgcn_cvt_pk_fp8_f32(clamp448(y0), clamp448(y1), p, false);
    p = __builtin_amdgcn_cvt_pk_fp8_f32(clamp448(y2), clamp448(y3), p, true);
#ifdef HW_FP8_CVT16
    v2h lo = __builtin_amdgcn_cvt_pk_f16_fp8((short)(p & 0xffff));          // v.l
    v2h hi = __builtin_amdgcn_cvt_pk_f16_fp8((short)((unsigned)p >> 16));   // v.h
    dst[base + 0] = lo[0];
    dst[base + 1] = lo[1];
    dst[base + 2] = hi[0];
    dst[base + 3] = hi[1];
#else
    v2f d0 = __builtin_amdgcn_cvt_pk_f32_fp8(p, false);
    v2f d1 = __builtin_amdgcn_cvt_pk_f32_fp8(p, true);
    dst[base + 0] = (_Float16)d0[0];
    dst[base + 1] = (_Float16)d0[1];
    dst[base + 2] = (_Float16)d1[0];
    dst[base + 3] = (_Float16)d1[1];
#endif
#else
    dst[base + 0] = (_Float16)e4m3_rne_sw(y0);
    dst[base + 1] = (_Float16)e4m3_rne_sw(y1);
    dst[base + 2] = (_Float16)e4m3_rne_sw(y2);
    dst[base + 3] = (_Float16)e4m3_rne_sw(y3);
#endif
}

// Load the 16 f32 a lane owns of a 32-wide K chunk: 8 at +0 and 8 at +16.
__device__ __forceinline__ void load16(const float* __restrict__ p, float* v) {
    float4 x0 = *(const float4*)(p);
    float4 x1 = *(const float4*)(p + 4);
    float4 x2 = *(const float4*)(p + 16);
    float4 x3 = *(const float4*)(p + 20);
    v[0]=x0.x;  v[1]=x0.y;  v[2]=x0.z;  v[3]=x0.w;
    v[4]=x1.x;  v[5]=x1.y;  v[6]=x1.z;  v[7]=x1.w;
    v[8]=x2.x;  v[9]=x2.y;  v[10]=x2.z; v[11]=x2.w;
    v[12]=x3.x; v[13]=x3.y; v[14]=x3.z; v[15]=x3.w;
}

// Load a 16-f16 fragment (8 at +0, 8 at +16 halfs): two b128 loads, no cvt.
__device__ __forceinline__ v16h load_frag_f16(const _Float16* __restrict__ p) {
    v8h lo = *(const v8h*)(p);
    v8h hi = *(const v8h*)(p + 16);
    return __builtin_shufflevector(lo, hi, 0, 1, 2, 3, 4, 5, 6, 7,
                                   8, 9, 10, 11, 12, 13, 14, 15);
}

// ---------------------------------------------------------------------------
// Kernel 1: initialize output with broadcast bias; zero the amax slot in ws.
// ---------------------------------------------------------------------------
__global__ void init_kernel(float* __restrict__ out, const float* __restrict__ bias,
                            unsigned* __restrict__ amax_bits) {
    if (blockIdx.x == 0 && threadIdx.x == 0) *amax_bits = 0u;
    int idx = blockIdx.x * blockDim.x + threadIdx.x;
    int stride = gridDim.x * blockDim.x;
    for (int i = idx; i < M_DIM * N_DIM; i += stride)
        out[i] = bias[i % N_DIM];
}

// ---------------------------------------------------------------------------
// Kernel 1b: one-time A f32 -> f16 conversion into workspace (A is reused by
// every N-block and K-split; converting once removes 16 cvts + 4 loads per
// GEMM inner iteration).
// ---------------------------------------------------------------------------
__global__ void convert_a_kernel(const float* __restrict__ A,
                                 _Float16* __restrict__ A16) {
    int idx = blockIdx.x * blockDim.x + threadIdx.x;
    int stride = gridDim.x * blockDim.x;
    for (int i = idx; i < M_DIM * K_DIM; i += stride)
        A16[i] = (_Float16)A[i];
}

// ---------------------------------------------------------------------------
// Kernel 2: per-tensor amax over |W| (201 MB stream), float4 loads,
// wave32 shuffle reduce, atomicMax on non-negative float bit patterns.
// ---------------------------------------------------------------------------
__global__ void amax_kernel(const float* __restrict__ W,
                            unsigned* __restrict__ amax_bits, long n4) {
    long idx = (long)blockIdx.x * blockDim.x + threadIdx.x;
    long stride = (long)gridDim.x * blockDim.x;
    const float4* W4 = (const float4*)W;
    float m = 0.0f;
    for (long i = idx; i < n4; i += stride) {
        float4 v = W4[i];
        m = fmaxf(m, fmaxf(fmaxf(fabsf(v.x), fabsf(v.y)),
                           fmaxf(fabsf(v.z), fabsf(v.w))));
    }
    for (int off = 16; off > 0; off >>= 1)
        m = fmaxf(m, __shfl_xor(m, off, 32));
    if ((threadIdx.x & 31) == 0)
        atomicMax(amax_bits, __float_as_uint(m));   // valid: m >= 0
}

// ---------------------------------------------------------------------------
// Kernel 3: fused quantize + GEMM (accumulate in fp8 "q-space", scale by the
// scalar recip in the epilogue: A @ (q*recip)^T == recip * (A @ q^T)).
//
// Block = 256 threads (8 wave32s); each wave handles a 16-wide N strip x M=32
// (two 16x16 C tiles). Grid = (N/128, KSPLIT); split-K partials accumulated
// with global_atomic_add_f32 into the bias-initialized output.
//
// WMMA 16-bit fragment striping per ISA 7.12.2:
//   lane<16  -> K = {0..7, 16..23}, lane>=16 -> K = {8..15, 24..31}.
// B column n of the 32x16 B tile is weight row (nBase + n), contiguous in K.
// ---------------------------------------------------------------------------
template <bool USE_A16>
__global__ void __launch_bounds__(256)
fp8_gemm_kernel(const float* __restrict__ A, const _Float16* __restrict__ A16,
                const float* __restrict__ W, float* __restrict__ out,
                const unsigned* __restrict__ amax_bits) {
    const int lane = threadIdx.x & 31;
    const int wave = threadIdx.x >> 5;
    const int l15  = lane & 15;
    const int half = lane >> 4;                 // 0: K lo pattern, 1: K hi pattern
    const int nBase = blockIdx.x * 128 + wave * 16;
    const int kStart = blockIdx.y * (K_DIM / KSPLIT);
    const int kEnd   = kStart + (K_DIM / KSPLIT);

    const float amax  = fmaxf(__uint_as_float(*amax_bits), 1e-12f);
    const float scale = FP8_MAXF / amax;
    const float recip = amax / FP8_MAXF;

    const float*    wrow   = W   + (size_t)(nBase + l15) * K_DIM + half * 8;
    const float*    arow0f = A   + (size_t)l15 * K_DIM + half * 8;
    const float*    arow1f = arow0f + (size_t)16 * K_DIM;
    const _Float16* arow0h = A16 + (size_t)l15 * K_DIM + half * 8;
    const _Float16* arow1h = arow0h + (size_t)16 * K_DIM;

    v8f c0 = {};
    v8f c1 = {};
    int p0 = 0, p1 = 0, p2 = 0, p3 = 0;         // persistent fp8 pack registers

#pragma unroll 2
    for (int k0 = kStart; k0 < kEnd; k0 += 32) {
        __builtin_prefetch(wrow + k0 + 64, 0, 1);   // global_prefetch_b8

        float wv[16];
        load16(wrow + k0, wv);

        v16h bf;
        e4m3_quad(wv[0] * scale,  wv[1] * scale,  wv[2] * scale,  wv[3] * scale,  p0, bf, 0);
        e4m3_quad(wv[4] * scale,  wv[5] * scale,  wv[6] * scale,  wv[7] * scale,  p1, bf, 4);
        e4m3_quad(wv[8] * scale,  wv[9] * scale,  wv[10] * scale, wv[11] * scale, p2, bf, 8);
        e4m3_quad(wv[12] * scale, wv[13] * scale, wv[14] * scale, wv[15] * scale, p3, bf, 12);

        v16h a0f, a1f;
        if constexpr (USE_A16) {
            a0f = load_frag_f16(arow0h + k0);
            a1f = load_frag_f16(arow1h + k0);
        } else {
            float a0v[16], a1v[16];
            load16(arow0f + k0, a0v);
            load16(arow1f + k0, a1v);
#pragma unroll
            for (int i = 0; i < 16; ++i) {
                a0f[i] = (_Float16)a0v[i];
                a1f[i] = (_Float16)a1v[i];
            }
        }

        c0 = __builtin_amdgcn_wmma_f32_16x16x32_f16(false, a0f, false, bf,
                                                    (short)0, c0, false, false);
        c1 = __builtin_amdgcn_wmma_f32_16x16x32_f16(false, a1f, false, bf,
                                                    (short)0, c1, false, false);
    }

    // C layout (32-bit 16x16): VGPR r -> M = r (lanes 0-15) / r+8 (lanes 16-31),
    // N = lane & 15. Apply the scalar dequant factor here.
    const int nCol = nBase + l15;
#pragma unroll
    for (int r = 0; r < 8; ++r) {
        const int m = half ? (r + 8) : r;
        atomicAdd(out + (size_t)m        * N_DIM + nCol, c0[r] * recip);
        atomicAdd(out + (size_t)(m + 16) * N_DIM + nCol, c1[r] * recip);
    }
}

// ---------------------------------------------------------------------------
extern "C" void kernel_launch(void* const* d_in, const int* in_sizes, int n_in,
                              void* d_out, int out_size, void* d_ws, size_t ws_size,
                              hipStream_t stream) {
    const float* A    = (const float*)d_in[0];   // [32, 4096]
    const float* W    = (const float*)d_in[1];   // [12288, 4096]
    const float* bias = (const float*)d_in[2];   // [12288]
    float* out        = (float*)d_out;           // [32, 12288]
    unsigned* amax_bits = (unsigned*)d_ws;
    _Float16* A16 = (_Float16*)((char*)d_ws + WS_A16_OFFSET);

    const size_t a16_bytes = (size_t)M_DIM * K_DIM * sizeof(_Float16);
    const bool use_a16 = ws_size >= WS_A16_OFFSET + a16_bytes;   // deterministic

    // 1) bias into out, zero amax accumulator
    init_kernel<<<(M_DIM * N_DIM + 255) / 256, 256, 0, stream>>>(out, bias, amax_bits);

    // 1b) one-time A -> f16 staging (if workspace permits)
    if (use_a16)
        convert_a_kernel<<<128, 256, 0, stream>>>(A, A16);

    // 2) per-tensor amax of |W|
    const long n4 = (long)N_DIM * K_DIM / 4;
    amax_kernel<<<1024, 256, 0, stream>>>(W, amax_bits, n4);

    // 3) fused quant + WMMA GEMM with split-K accumulation
    dim3 grid(N_DIM / 128, KSPLIT);
    if (use_a16)
        fp8_gemm_kernel<true><<<grid, 256, 0, stream>>>(A, A16, W, out, amax_bits);
    else
        fp8_gemm_kernel<false><<<grid, 256, 0, stream>>>(A, A16, W, out, amax_bits);
}
